// SPCSA_3015067042105
// MI455X (gfx1250) — compile-verified, matrix operand order checked by reading, not run on hardware
//
#include <hip/hip_runtime.h>
#include <math.h>

typedef __attribute__((ext_vector_type(16))) __bf16 v16bf;
typedef __attribute__((ext_vector_type(8)))  float  v8f;
typedef __attribute__((ext_vector_type(4))) unsigned int v4u;
typedef __attribute__((ext_vector_type(8))) int v8i;
typedef __attribute__((ext_vector_type(4))) int v4i;

#define B_     8
#define HW_    4096
#define NTOT   32768      // B * H * W
#define HEADS_ 8

// CDNA5 16-bit A/B fragment K mapping (ISA 7.12.2):
// lanes 0-15:  elems 0..7 -> K 0..7,  elems 8..15 -> K 16..23
// lanes 16-31: elems 0..7 -> K 8..15, elems 8..15 -> K 24..31
__device__ __forceinline__ int kmap(int lane, int e) {
  int h0 = (lane & 16) ? 8 : 0;
  return (e < 8) ? (h0 + e) : (16 + h0 + (e - 8));
}

// Low 32 bits of a generic pointer to LDS == LDS byte offset (flat aperture).
__device__ __forceinline__ unsigned lds_off(const void* p) {
  return (unsigned)(unsigned long long)(uintptr_t)p;
}

// ---------------------------------------------------------------------------
// Tensor Data Mover: 2D tile load global -> LDS (ISA ch.8, D# descriptor).
//   tile_w  : elements per row (dim0), tile_h: rows (dim1)
//   stride  : row stride in elements
//   ds_code : 1 = 2-byte elems, 2 = 4-byte elems
//   pad_int : pad interval code (dwords before pad = 2<<code)
//   pad_amt : pad amount code (dwords inserted = code+1)
// ---------------------------------------------------------------------------
__device__ __forceinline__ void tdm_load_2d(unsigned lds_byte_off, const void* gptr,
                                            unsigned tile_w, unsigned tile_h,
                                            unsigned long long stride_elems,
                                            unsigned ds_code, unsigned pad_int,
                                            unsigned pad_amt)
{
  unsigned long long ga = (unsigned long long)(uintptr_t)gptr;
  v4u g0;
  g0.x = 1u;                                                   // count=1 (user D#)
  g0.y = lds_byte_off;                                         // lds_addr
  g0.z = (unsigned)ga;                                         // global_addr lo
  g0.w = ((unsigned)(ga >> 32) & 0x01ffffffu) | (2u << 30);    // global_addr hi | type=2
  const unsigned td0 = 0x40000000u, td1 = 0x40000000u;         // huge dims: no OOB clip
  v8i g1;
  g1[0] = (int)((ds_code << 16) | (1u << 20) | (pad_int << 22) | (pad_amt << 25));
  g1[1] = (int)((td0 & 0xffffu) << 16);                        // tensor_dim0[15:0]
  g1[2] = (int)((td0 >> 16) | ((td1 & 0xffffu) << 16));        // td0 hi | td1 lo
  g1[3] = (int)((td1 >> 16) | (tile_w << 16));                 // td1 hi | tile_dim0
  g1[4] = (int)(tile_h & 0xffffu);                             // tile_dim1 | tile_dim2=0
  g1[5] = (int)(unsigned)(stride_elems & 0xffffffffull);       // dim0_stride lo32
  g1[6] = (int)(unsigned)((stride_elems >> 32) & 0xffffull);   // dim0_stride hi16
  g1[7] = 0;
  v4i z4 = {0, 0, 0, 0};
#if __clang_major__ >= 23
  v8i z8 = {0, 0, 0, 0, 0, 0, 0, 0};
  __builtin_amdgcn_tensor_load_to_lds(g0, g1, z4, z4, z8, 0);
#else
  __builtin_amdgcn_tensor_load_to_lds(g0, g1, z4, z4, 0);
#endif
}

// ---------------------------------------------------------------------------
// bf16 WMMA GEMM: Y[M,NTOT] = W[M,K] @ X[K,NTOT] (+bias)(+relu)
// 256 thr (8 waves), 64x64 tile, K-chunk 32, TDM double-buffered staging.
// ---------------------------------------------------------------------------
__global__ __launch_bounds__(256)
void gemm_bf16_wmma(const float* __restrict__ W, const float* __restrict__ X,
                    const float* __restrict__ bias, float* __restrict__ Y,
                    int M, int K, int x_bchw, int relu)
{
  __shared__ float sW[2][64][34];   // rows of 32 f32, TDM-padded +2 dwords
  __shared__ float sX[2][32][66];   // rows of 64 f32, TDM-padded +2 dwords
  const int t = threadIdx.x, lane = t & 31, w = t >> 5;
  const int n0 = blockIdx.x * 64, m0 = blockIdx.y * 64;
  const int mi = w >> 1, ni0 = (w & 1) * 2;
  v8f c0 = {}, c1 = {};
  const int nch = K >> 5;

  auto issue = [&](int ci, int buf) {
    int k0 = ci * 32;
    tdm_load_2d(lds_off(&sW[buf][0][0]), W + (size_t)m0 * K + k0,
                32, 64, (unsigned long long)K, 2, 4, 1);        // 32 dw rows, +2 dw pad
    size_t xoff;
    unsigned long long xstride;
    if (x_bchw) {
      int b = n0 >> 12, p = n0 & 4095;
      xoff = ((size_t)b * K + k0) * (size_t)HW_ + p;
      xstride = HW_;
    } else {
      xoff = (size_t)k0 * NTOT + n0;
      xstride = NTOT;
    }
    tdm_load_2d(lds_off(&sX[buf][0][0]), X + xoff,
                64, 32, xstride, 2, 5, 1);                      // 64 dw rows, +2 dw pad
  };

  if (w == 0) issue(0, 0);
  for (int ci = 0; ci < nch; ++ci) {
    int buf = ci & 1;
    if (w == 0) {
      if (ci + 1 < nch) { issue(ci + 1, buf ^ 1); __builtin_amdgcn_s_wait_tensorcnt(2); }
      else              { __builtin_amdgcn_s_wait_tensorcnt(0); }
    }
    __syncthreads();
    int arow  = mi * 16 + (lane & 15);
    int brow0 = ni0 * 16 + (lane & 15);
    v16bf a, b0, b1;
#pragma unroll
    for (int e = 0; e < 16; ++e) {
      int kk = kmap(lane, e);
      a[e]  = (__bf16)sW[buf][arow][kk];
      b0[e] = (__bf16)sX[buf][kk][brow0];
      b1[e] = (__bf16)sX[buf][kk][brow0 + 16];
    }
    c0 = __builtin_amdgcn_wmma_f32_16x16x32_bf16(false, a, false, b0, (short)0, c0, false, false);
    c1 = __builtin_amdgcn_wmma_f32_16x16x32_bf16(false, a, false, b1, (short)0, c1, false, false);
    __syncthreads();
  }

  const int hisel = (lane >> 4) << 3;
#pragma unroll
  for (int r = 0; r < 8; ++r) {
    int m = m0 + mi * 16 + r + hisel;
    float bv = bias ? bias[m] : 0.f;
    float v0 = c0[r] + bv, v1 = c1[r] + bv;
    if (relu) { v0 = fmaxf(v0, 0.f); v1 = fmaxf(v1, 0.f); }
    int nl0 = ni0 * 16 + (lane & 15);
    Y[(size_t)m * NTOT + (n0 + nl0)]      = v0;
    Y[(size_t)m * NTOT + (n0 + nl0 + 16)] = v1;
  }
}

// ---------------------------------------------------------------------------
// Depthwise 3x3, SAME pad, fp32 in -> bf16 out. Layout [ch][b*4096+p].
// ---------------------------------------------------------------------------
__global__ __launch_bounds__(256)
void dwconv3x3(const float* __restrict__ in, const float* __restrict__ wdw,
               __bf16* __restrict__ out)
{
  size_t gid = (size_t)blockIdx.x * 256 + threadIdx.x;
  int ch  = (int)(gid >> 15);
  int rem = (int)(gid & 32767);
  int b = rem >> 12, p = rem & 4095;
  int y = p >> 6,    x = p & 63;
  const float* wp   = wdw + (size_t)ch * 9;
  const float* base = in + (size_t)ch * NTOT + (size_t)b * HW_;
  float acc = 0.f;
#pragma unroll
  for (int dy = -1; dy <= 1; ++dy) {
    int yy = y + dy;
    if (yy < 0 || yy > 63) continue;
#pragma unroll
    for (int dx = -1; dx <= 1; ++dx) {
      int xx = x + dx;
      if (xx < 0 || xx > 63) continue;
      acc += wp[(dy + 1) * 3 + (dx + 1)] * base[yy * 64 + xx];
    }
  }
  out[gid] = (__bf16)acc;
}

// ---------------------------------------------------------------------------
// Row L2-norm scale for q,k rows (raw dw output). scale[ch*8+b].
// ---------------------------------------------------------------------------
__global__ __launch_bounds__(256)
void rownorm(const __bf16* __restrict__ qk, float* __restrict__ scale)
{
  __shared__ float sred[256];
  int row = blockIdx.x;
  int ch = row >> 3, b = row & 7;
  const __bf16* base = qk + (size_t)ch * NTOT + (size_t)b * HW_;
  float s = 0.f;
  for (int i = threadIdx.x; i < HW_; i += 256) {
    float v = (float)base[i];
    s += v * v;
  }
  sred[threadIdx.x] = s;
  __syncthreads();
  for (int off = 128; off > 0; off >>= 1) {
    if (threadIdx.x < off) sred[threadIdx.x] += sred[threadIdx.x + off];
    __syncthreads();
  }
  if (threadIdx.x == 0)
    scale[row] = 1.f / fmaxf(sqrtf(sred[0]), 1e-12f);
}

// ---------------------------------------------------------------------------
// Gate: dynamic_k = floor(64 * mean(sigmoid(wg2 . relu_g1 + bg2)))
// ---------------------------------------------------------------------------
__global__ void gate_init(float* s) { s[0] = 0.f; }

__global__ __launch_bounds__(256)
void gate_reduce(const float* __restrict__ g1, const float* __restrict__ wg2,
                 const float* __restrict__ bg2, float* __restrict__ accum)
{
  __shared__ float sred[256];
  int n = blockIdx.x * 256 + threadIdx.x;
  float acc = bg2[0];
  for (int c = 0; c < 256; ++c) acc += wg2[c] * g1[(size_t)c * NTOT + n];
  sred[threadIdx.x] = 1.f / (1.f + expf(-acc));
  __syncthreads();
  for (int off = 128; off > 0; off >>= 1) {
    if (threadIdx.x < off) sred[threadIdx.x] += sred[threadIdx.x + off];
    __syncthreads();
  }
  if (threadIdx.x == 0) atomicAdd(accum, sred[0]);
}

__global__ void gate_final(float* s)
{
  float mean = s[0] / (float)NTOT;
  s[1] = fminf(fmaxf(floorf(64.f * mean), 0.f), 64.f);
}

// ---------------------------------------------------------------------------
// Attention: one block per (b, head); 512 threads = 16 waves.
// Phase 1: raw q@k^T via WMMA (TDM double-buffered 64x128 chunks); L2-norm
//          scales + temperature folded at the epilogue (exact algebra).
// Phase 2: rank-based top-dynamic_k mask + softmax.
// Phase 3: P@V via WMMA, V staged by TDM in 64x64 LDS tiles (double-buffered).
// ---------------------------------------------------------------------------
#define KC 128
__global__ __launch_bounds__(512)
void attn_kernel(const __bf16* __restrict__ qkv, const float* __restrict__ scale,
                 const float* __restrict__ temperature, const float* __restrict__ scal,
                 const float* __restrict__ a1, const float* __restrict__ a2,
                 const float* __restrict__ a3, const float* __restrict__ a4,
                 float* __restrict__ out)
{
  __shared__ __bf16 sQ[2][64][132];   // rows of 128 bf16 (64 dw), +2 dw pad
  __shared__ __bf16 sKt[2][64][132];
  __shared__ float  sA[64][65];
  __shared__ __bf16 sP[64][66];
  __shared__ __bf16 sV[2][64][68];    // rows of 64 bf16 (32 dw), +2 dw pad

  const int bb   = blockIdx.x >> 3;
  const int h    = blockIdx.x & 7;
  const int t    = threadIdx.x;
  const int lane = t & 31;
  const int w    = t >> 5;               // 0..15
  const int mi   = w >> 2, ni = w & 3;

  const __bf16* qbase = qkv + ((size_t)(h * 64)        * NTOT) + (size_t)bb * HW_;
  const __bf16* kbase = qkv + ((size_t)(512 + h * 64)  * NTOT) + (size_t)bb * HW_;
  const __bf16* vbase = qkv + ((size_t)(1024 + h * 64) * NTOT) + (size_t)bb * HW_;

  auto issueQK = [&](int ci, int buf) {
    tdm_load_2d(lds_off(&sQ[buf][0][0]),  qbase + ci * KC, KC, 64, NTOT, 1, 5, 1);
    tdm_load_2d(lds_off(&sKt[buf][0][0]), kbase + ci * KC, KC, 64, NTOT, 1, 5, 1);
  };

  v8f acc = {};
  const int NCH = HW_ / KC;              // 32
  if (w == 0) issueQK(0, 0);
  for (int ci = 0; ci < NCH; ++ci) {
    int buf = ci & 1;
    if (w == 0) {
      if (ci + 1 < NCH) { issueQK(ci + 1, buf ^ 1); __builtin_amdgcn_s_wait_tensorcnt(2); }
      else              { __builtin_amdgcn_s_wait_tensorcnt(0); }
    }
    __syncthreads();
    int arow = mi * 16 + (lane & 15);
    int brow = ni * 16 + (lane & 15);
#pragma unroll
    for (int kc = 0; kc < KC / 32; ++kc) {
      v16bf a, b;
#pragma unroll
      for (int e = 0; e < 16; ++e) {
        int kk = kc * 32 + kmap(lane, e);
        a[e] = sQ[buf][arow][kk];
        b[e] = sKt[buf][brow][kk];
      }
      acc = __builtin_amdgcn_wmma_f32_16x16x32_bf16(false, a, false, b, (short)0, acc, false, false);
    }
    __syncthreads();
  }

  // epilogue: fold q/k L2-norm scales + temperature, spill attn to LDS
  const float temp  = temperature[h];
  const int   hisel = (lane >> 4) << 3;
  const float skv   = scale[(512 + h * 64 + ni * 16 + (lane & 15)) * 8 + bb];
#pragma unroll
  for (int r = 0; r < 8; ++r) {
    int m = mi * 16 + r + hisel;
    float sqv = scale[(h * 64 + m) * 8 + bb];
    sA[m][ni * 16 + (lane & 15)] = acc[r] * sqv * skv * temp;
  }
  __syncthreads();

  // phase 2: dynamic top-k + softmax, one thread per attn row
  if (t < 64) {
    int dk = (int)scal[1];
    if (dk < 0) dk = 0;
    if (dk > 64) dk = 64;
    float m = -INFINITY;
    unsigned long long keep = 0ull;
    for (int i = 0; i < 64; ++i) {
      float ai = sA[t][i];
      int rank = 0;
      for (int j = 0; j < 64; ++j) {
        float aj = sA[t][j];
        rank += (aj > ai) || (aj == ai && j < i);   // stable argsort tie-break
      }
      if (rank < dk) { keep |= (1ull << i); m = fmaxf(m, ai); }
    }
    float denom = 0.f;
    for (int i = 0; i < 64; ++i)
      if ((keep >> i) & 1) denom += expf(sA[t][i] - m);
    float inv = (denom > 0.f) ? (1.f / denom) : 0.f;
    for (int i = 0; i < 64; ++i) {
      float p = ((keep >> i) & 1) ? expf(sA[t][i] - m) * inv : 0.f;
      sP[t][i] = (__bf16)p;
    }
  }
  __syncthreads();

  // phase 3: out = P @ V * (a1+a2+a3+a4); V staged 64x64 via TDM
  const float ascale = a1[0] + a2[0] + a3[0] + a4[0];
  const int   ct = w & 3, ns = w >> 2;
  const int   prow = ct * 16 + (lane & 15);
  v16bf pa0, pa1;
#pragma unroll
  for (int e = 0; e < 16; ++e) {
    int kk = kmap(lane, e);
    pa0[e] = sP[prow][kk];
    pa1[e] = sP[prow][kk + 32];
  }
  auto issueV = [&](int nb, int buf) {
    tdm_load_2d(lds_off(&sV[buf][0][0]), vbase + nb * 64, 64, 64, NTOT, 1, 4, 1);
  };
  if (w == 0) issueV(0, 0);
  for (int nb = 0; nb < 64; ++nb) {
    int buf = nb & 1;
    if (w == 0) {
      if (nb + 1 < 64) { issueV(nb + 1, buf ^ 1); __builtin_amdgcn_s_wait_tensorcnt(1); }
      else             { __builtin_amdgcn_s_wait_tensorcnt(0); }
    }
    __syncthreads();
    int ncl = ns * 16 + (lane & 15);
    v16bf vb0, vb1;
#pragma unroll
    for (int e = 0; e < 16; ++e) {
      int d = kmap(lane, e);
      vb0[e] = sV[buf][d][ncl];
      vb1[e] = sV[buf][d + 32][ncl];
    }
    v8f o = {};
    o = __builtin_amdgcn_wmma_f32_16x16x32_bf16(false, pa0, false, vb0, (short)0, o, false, false);
    o = __builtin_amdgcn_wmma_f32_16x16x32_bf16(false, pa1, false, vb1, (short)0, o, false, false);
    int ncol = nb * 64 + ncl;
#pragma unroll
    for (int r = 0; r < 8; ++r) {
      int cidx = ct * 16 + r + hisel;
      out[((size_t)bb * 512 + h * 64 + cidx) * (size_t)HW_ + ncol] = o[r] * ascale;
    }
    __syncthreads();
  }
}

// ---------------------------------------------------------------------------
extern "C" void kernel_launch(void* const* d_in, const int* in_sizes, int n_in,
                              void* d_out, int out_size, void* d_ws, size_t ws_size,
                              hipStream_t stream)
{
  const float* x    = (const float*)d_in[0];
  const float* w0   = (const float*)d_in[1];
  const float* b0   = (const float*)d_in[2];
  const float* wqkv = (const float*)d_in[3];
  const float* wdw  = (const float*)d_in[4];
  const float* temp = (const float*)d_in[5];
  const float* wg1  = (const float*)d_in[6];
  const float* bg1  = (const float*)d_in[7];
  const float* wg2  = (const float*)d_in[8];
  const float* bg2  = (const float*)d_in[9];
  const float* a1   = (const float*)d_in[10];
  const float* a2   = (const float*)d_in[11];
  const float* a3   = (const float*)d_in[12];
  const float* a4   = (const float*)d_in[13];
  float* ws = (float*)d_ws;

  // workspace layout (float units)
  const size_t OFF_SCAL  = 0;                               // [0]=gate sum, [1]=dynamic_k
  const size_t OFF_SCALE = 16;                              // 8192 norm scales
  const size_t OFF_X1    = 8224;                            // 512 x 32768 fp32
  const size_t OFF_QKV1  = OFF_X1 + (size_t)512 * NTOT;     // 1536 x 32768 fp32
  const size_t OFF_G1    = OFF_QKV1 + (size_t)1536 * NTOT;  // 256 x 32768 fp32
  const size_t OFF_QKVDW = OFF_G1;                          // 1536 x 32768 bf16 (reuses g1)

  float*  scal    = ws + OFF_SCAL;
  float*  scaleqk = ws + OFF_SCALE;
  float*  x1      = ws + OFF_X1;
  float*  qkv1    = ws + OFF_QKV1;
  float*  g1      = ws + OFF_G1;
  __bf16* qkvdw   = (__bf16*)(ws + OFF_QKVDW);

  // 1) x1 = w0 @ x + b0   (1x1 conv as GEMM, input is BCHW)
  gemm_bf16_wmma<<<dim3(NTOT / 64, 512 / 64), 256, 0, stream>>>(w0, x, b0, x1, 512, 512, 1, 0);
  // 2) gate branch (finishes before depthwise overwrites g1 region)
  gemm_bf16_wmma<<<dim3(NTOT / 64, 256 / 64), 256, 0, stream>>>(wg1, x1, bg1, g1, 256, 512, 0, 1);
  gate_init<<<1, 1, 0, stream>>>(scal);
  gate_reduce<<<NTOT / 256, 256, 0, stream>>>(g1, wg2, bg2, scal);
  gate_final<<<1, 1, 0, stream>>>(scal);
  // 3) qkv = wqkv @ x1 (bias-free)
  gemm_bf16_wmma<<<dim3(NTOT / 64, 1536 / 64), 256, 0, stream>>>(wqkv, x1, nullptr, qkv1, 1536, 512, 0, 0);
  // 4) depthwise 3x3 -> bf16
  dwconv3x3<<<(1536 * NTOT) / 256, 256, 0, stream>>>(qkv1, wdw, qkvdw);
  // 5) q,k row norms (raw; folded into attention epilogue)
  rownorm<<<8192, 256, 0, stream>>>(qkvdw, scaleqk);
  // 6) attention per (b, head)
  attn_kernel<<<B_ * HEADS_, 512, 0, stream>>>(qkvdw, scaleqk, temp, scal,
                                               a1, a2, a3, a4, (float*)d_out);
}
